// TTT_1992864825518
// MI455X (gfx1250) — matrix operand order repaired and checked by reading, not compiled, not run on hardware
//
#include <hip/hip_runtime.h>

// CDNA5 / gfx1250: wave32, WMMA 16x16x32 bf16 -> f32 accumulate.
typedef __attribute__((ext_vector_type(16))) __bf16 v16bf;
typedef __attribute__((ext_vector_type(4)))  __bf16 v4bf;
typedef __attribute__((ext_vector_type(8)))  float  v8f;

namespace {
constexpr int   kB  = 16;
constexpr int   kS  = 2048;
constexpr int   kD  = 512;
constexpr float kLR = 0.01f;
// Wn = Wc + (-LR * 2/(B*D)) * (y-x)^T tv ; fold the whole coefficient into e.
constexpr float kCoef = -kLR * 2.0f / (float)(kB * kD);
}

// One workgroup per 16-feature slice of W (j-tile). W-row updates are
// self-contained per feature row (rank-16 outer product), so the S-scan is
// embarrassingly parallel across j. Each of the 8 waves owns a 64-wide K
// slice of W^T, kept in fp32 VGPRs (WMMA C/D layout) for the entire scan,
// with a bf16 mirror in WMMA B-operand layout for the forward/readout GEMMs.
__launch_bounds__(256, 1)
__global__ void ttt_scan_wmma(const float* __restrict__ x_g,
                              const float* __restrict__ n_g,
                              const float* __restrict__ w_g,
                              float* __restrict__ out_g) {
  __shared__ __bf16 sTb[kB * kD];   // tv_s tile [b][k], bf16        (16 KB)
  __shared__ __bf16 sTt[kD * kB];   // tv_s^T tile [k][b], bf16      (16 KB)
  __shared__ __bf16 sXb[kB * kD];   // x_s tile [b][k], bf16         (16 KB)
  __shared__ float  sXj[kB * 16];   // x_s[:, j0:j0+16], fp32        (1 KB)
  __shared__ float  sP[8 * 256];    // per-wave y/readout partials   (8 KB)
  __shared__ __bf16 sMb[8 * 256];   // per-wave bf16 mirror scratch  (4 KB)
  __shared__ __bf16 sE[16 * 16];    // e = coef*(y-x) tile, bf16     (0.5 KB)

  const int tid = threadIdx.x;
  const int w   = tid >> 5;        // wave id 0..7 (owns K range [64w, 64w+64))
  const int l   = tid & 31;        // lane
  const int jc  = l & 15;          // N (feature col within tile) for C/D tiles
  const int g   = l >> 4;          // lane half
  const int m   = l & 15;          // M row for A operands
  const int j0  = blockIdx.x * 16; // global feature base of this workgroup
  const int kw  = w * 64;          // wave's K base

  // 16-bit A-operand layout: halfword h of the v16bf <-> K index (per lane half g).
  int kofh[16];
#pragma unroll
  for (int h = 0; h < 16; ++h) {
    const int v = h >> 1, p = h & 1;
    kofh[h] = (v < 4) ? (2 * v + p + 8 * g) : (16 + 2 * (v - 4) + p + 8 * g);
  }

  // ---- load persistent W^T slice into fp32 WMMA C/D-layout registers ----
  // chunk c holds Wt rows [kw+16c, kw+16c+16), cols [j0, j0+16):
  // element (k', j): vgpr r = k'%8, lane = j + 16*(k'/8).
  v8f wt[4];
#pragma unroll
  for (int c = 0; c < 4; ++c)
#pragma unroll
    for (int r = 0; r < 8; ++r) {
      const int k = kw + c * 16 + r + 8 * g;
      wt[c][r] = w_g[(size_t)(j0 + jc) * kD + k];
    }

  // Rebuild bf16 mirror (B-operand layout: lane%16 = N,
  // halfword h = K kofh[h]) from the fp32 master via a per-wave bf16 LDS roundtrip:
  // convert once on the write side, read back raw bf16 (strided 16-bit loads).
  // Same-wave DS ops are in-order, so no barrier is needed.
  auto mirror_half = [&](int q) -> v16bf {
    v16bf wb;
#pragma unroll
    for (int cc = 0; cc < 2; ++cc) {
      const v8f ch = wt[q * 2 + cc];
#pragma unroll
      for (int r = 0; r < 8; ++r)
        sMb[w * 256 + (r + 8 * g) * 16 + jc] = (__bf16)ch[r];
#pragma unroll
      for (int h = 0; h < 16; ++h)
        if ((kofh[h] >> 4) == cc)
          wb[h] = sMb[w * 256 + (kofh[h] & 15) * 16 + jc];
    }
    return wb;
  };

  v16bf wb0 = mirror_half(0);
  v16bf wb1 = mirror_half(1);
  __syncthreads();

  for (int s = 0; s < kS; ++s) {
    // ---- stage x / tv tiles: float4 global loads, packed bf16 LDS stores ----
    for (int i = tid * 4; i < kB * kD; i += 256 * 4) {
      const int b = i >> 9, k = i & (kD - 1);
      const size_t gi = (size_t)b * kS * kD + (size_t)s * kD + k;
      const float4 xv = *(const float4*)&x_g[gi];
      const float4 nv = *(const float4*)&n_g[gi];
      const float4 tv = make_float4(xv.x + nv.x, xv.y + nv.y,
                                    xv.z + nv.z, xv.w + nv.w);
      v4bf xb, tb;
      xb[0] = (__bf16)xv.x; xb[1] = (__bf16)xv.y;
      xb[2] = (__bf16)xv.z; xb[3] = (__bf16)xv.w;
      tb[0] = (__bf16)tv.x; tb[1] = (__bf16)tv.y;
      tb[2] = (__bf16)tv.z; tb[3] = (__bf16)tv.w;
      *(v4bf*)&sXb[i] = xb;
      *(v4bf*)&sTb[i] = tb;
#pragma unroll
      for (int t = 0; t < 4; ++t) sTt[(k + t) * kB + b] = tb[t];
      if (k >= j0 && k < j0 + 16)   // 4-groups are fully inside/outside
        *(float4*)&sXj[b * 16 + (k - j0)] = xv;
      if (s + 1 < kS) {
        __builtin_prefetch(&x_g[gi + kD], 0, 1);
        __builtin_prefetch(&n_g[gi + kD], 0, 1);
      }
    }
    __syncthreads();

    // ---- phase A: partial y = tv x Wt over this wave's K slice (2 WMMAs) ----
    v8f acc = {0.f, 0.f, 0.f, 0.f, 0.f, 0.f, 0.f, 0.f};
#pragma unroll
    for (int q = 0; q < 2; ++q) {
      v16bf a;
#pragma unroll
      for (int h = 0; h < 16; ++h)
        a[h] = sTb[m * kD + kw + q * 32 + kofh[h]];
      acc = __builtin_amdgcn_wmma_f32_16x16x32_bf16(
          false, a, false, (q ? wb1 : wb0), (short)0, acc, false, false);
    }
#pragma unroll
    for (int r = 0; r < 8; ++r)
      sP[w * 256 + (r + 8 * g) * 16 + jc] = acc[r];
    __syncthreads();

    // ---- reduce 8 partials -> e = coef*(y - x) (one element per thread) ----
    {
      const int b = tid >> 4, j = tid & 15;
      float y = 0.f;
#pragma unroll
      for (int ww = 0; ww < 8; ++ww) y += sP[ww * 256 + b * 16 + j];
      sE[b * 16 + j] = (__bf16)(kCoef * (y - sXj[b * 16 + j]));
    }
    __syncthreads();

    // ---- phase B: Wt += tv^T x e  (rank-16 update, K zero-padded to 32) ----
    v16bf eb;
#pragma unroll
    for (int h = 0; h < 16; ++h)
      eb[h] = (kofh[h] < 16) ? sE[kofh[h] * 16 + (l & 15)] : (__bf16)0.0f;
#pragma unroll
    for (int c = 0; c < 4; ++c) {
      v16bf a;  // A = tv^T subtile: (M=k'rel, K=b), b >= 16 padded with zero
      const int kk = kw + c * 16 + m;
#pragma unroll
      for (int h = 0; h < 16; ++h) {
        const int K = kofh[h];
        a[h] = (K < 16) ? sTt[kk * kB + K] : (__bf16)0.0f;
      }
      wt[c] = __builtin_amdgcn_wmma_f32_16x16x32_bf16(
          false, a, false, eb, (short)0, wt[c], false, false);
    }
    wb0 = mirror_half(0);
    wb1 = mirror_half(1);

    // ---- phase C: partial readout = x @ Wt_new over this wave's K slice ----
    v8f acc2 = {0.f, 0.f, 0.f, 0.f, 0.f, 0.f, 0.f, 0.f};
#pragma unroll
    for (int q = 0; q < 2; ++q) {
      v16bf a;
#pragma unroll
      for (int h = 0; h < 16; ++h)
        a[h] = sXb[m * kD + kw + q * 32 + kofh[h]];
      acc2 = __builtin_amdgcn_wmma_f32_16x16x32_bf16(
          false, a, false, (q ? wb1 : wb0), (short)0, acc2, false, false);
    }
#pragma unroll
    for (int r = 0; r < 8; ++r)
      sP[w * 256 + (r + 8 * g) * 16 + jc] = acc2[r];
    __syncthreads();

    // ---- reduce + store readout tile ----
    {
      const int b = tid >> 4, j = tid & 15;
      float y = 0.f;
#pragma unroll
      for (int ww = 0; ww < 8; ++ww) y += sP[ww * 256 + b * 16 + j];
      out_g[(size_t)b * kS * kD + (size_t)s * kD + j0 + j] = y;
    }
    __syncthreads();  // protect staged tiles before next step's staging
  }
}

extern "C" void kernel_launch(void* const* d_in, const int* in_sizes, int n_in,
                              void* d_out, int out_size, void* d_ws, size_t ws_size,
                              hipStream_t stream) {
  (void)in_sizes; (void)n_in; (void)out_size; (void)d_ws; (void)ws_size;
  const float* x = (const float*)d_in[0];  // in_seq [B,S,D] fp32
  const float* n = (const float*)d_in[1];  // noise  [B,S,D] fp32
  const float* W = (const float*)d_in[2];  // W      [D,D]   fp32
  float* out = (float*)d_out;              // [B,S,D] fp32
  ttt_scan_wmma<<<dim3(kD / 16), dim3(256), 0, stream>>>(x, n, W, out);
}
